// drugcell_nn_from_lee_64115271794768
// MI455X (gfx1250) — compile-verified, matrix operand order checked by reading, not run on hardware
//
#include <hip/hip_runtime.h>
#include <hip/hip_bf16.h>

// ---------------------------------------------------------------------------
// DrugCell-style network for MI455X (gfx1250, wave32, WMMA).
// Dominant work: B=16 batched (2048x2048)@(2048x32) with the 2048x2048 matrix
// generated on the fly via exp(), and a 512x2048 @ 2048x1024 GEMM.
// Both run on v_wmma_f32_16x16x32_f16. The tiny H=6 tail runs on VALU kernels.
// ---------------------------------------------------------------------------

typedef __attribute__((ext_vector_type(16))) _Float16 v16h;
typedef __attribute__((ext_vector_type(8)))  _Float16 v8h;
typedef __attribute__((ext_vector_type(8)))  float    v8f;

#define Bq   16
#define Gq   2048
#define Dq   2048
#define Eq   32
#define NLq  16
#define NMq  4
#define Hq   6
#define GSq  64

// ---------------------------------------------------------------------------
// elementwise f32 -> f16 conversion (weight staging)
// ---------------------------------------------------------------------------
__global__ void k_f2h(const float* __restrict__ in, _Float16* __restrict__ out, int n) {
    int i = blockIdx.x * 256 + threadIdx.x;
    if (i < n) out[i] = (_Float16)in[i];
}

// ---------------------------------------------------------------------------
// Kernel 1: gf[b*E+e][i] = (half)( sum_j exp(-0.5*(g_bi-g_bj)^2) * W_rbf[e][j]
//                                  + b_rbf[e] + emb[table[idx_b][i]][e] )
// One wave = one 16-row M-tile of one batch; A-tile built on the fly via exp.
// grid = (G/128, B), block = 256 (8 waves).
// ---------------------------------------------------------------------------
__global__ __launch_bounds__(256) void k_rbf_gv(
    const float* __restrict__ x,        // [B][G+D+1]
    const _Float16* __restrict__ Wh,    // [E][G]  (half)
    const float* __restrict__ b_rbf,    // [E]
    const float* __restrict__ emb,      // [G][E]
    const int* __restrict__ table,      // [NDRUG][G]
    _Float16* __restrict__ gf)          // [B*E][G] (half)
{
    __shared__ float gbuf[Gq];
    const int b = blockIdx.y;
    const int xstride = Gq + Dq + 1;
    for (int i = threadIdx.x; i < Gq; i += 256) gbuf[i] = x[(size_t)b * xstride + i];
    __syncthreads();
    const int idx  = (int)x[(size_t)b * xstride + Gq + Dq];
    const int wave = threadIdx.x >> 5;
    const int lane = threadIdx.x & 31;
    const int lm   = lane & 15;
    const int hi   = lane >> 4;
    const int base = hi * 8;
    const int itile = blockIdx.x * 128 + wave * 16;
    const float gi = gbuf[itile + lm];

    v8f acc0 = {}; v8f acc1 = {};
    const _Float16* w0 = Wh + (size_t)lm * Gq;        // B col e = lm
    const _Float16* w1 = Wh + (size_t)(lm + 16) * Gq; // B col e = lm+16

    for (int k0 = 0; k0 < Gq; k0 += 32) {
        v16h a, bm0, bm1;
        #pragma unroll
        for (int t = 0; t < 8; ++t) {
            float d0 = gi - gbuf[k0 + base + t];
            float d1 = gi - gbuf[k0 + 16 + base + t];
            a[t]     = (_Float16)__expf(-0.5f * d0 * d0);
            a[t + 8] = (_Float16)__expf(-0.5f * d1 * d1);
        }
        const v8h* p0 = (const v8h*)(w0 + k0 + base);
        const v8h* p1 = (const v8h*)(w1 + k0 + base);
        v8h b0lo = p0[0], b0hi = p0[2];   // +16 halves
        v8h b1lo = p1[0], b1hi = p1[2];
        #pragma unroll
        for (int t = 0; t < 8; ++t) {
            bm0[t] = b0lo[t]; bm0[t + 8] = b0hi[t];
            bm1[t] = b1lo[t]; bm1[t + 8] = b1hi[t];
        }
        acc0 = __builtin_amdgcn_wmma_f32_16x16x32_f16(false, a, false, bm0, (short)0, acc0, false, false);
        acc1 = __builtin_amdgcn_wmma_f32_16x16x32_f16(false, a, false, bm1, (short)0, acc1, false, false);
    }
    // epilogue: lane holds col e = lm (+16), rows i = istart + r (contiguous)
    const int e0 = lm, e1 = lm + 16;
    const float bb0 = b_rbf[e0], bb1 = b_rbf[e1];
    const int istart = itile + hi * 8;
    v8h o0, o1;
    #pragma unroll
    for (int r = 0; r < 8; ++r) {
        int ig = istart + r;
        int trow = table[(size_t)idx * Gq + ig];
        o0[r] = (_Float16)(acc0[r] + bb0 + emb[(size_t)trow * Eq + e0]);
        o1[r] = (_Float16)(acc1[r] + bb1 + emb[(size_t)trow * Eq + e1]);
    }
    *(v8h*)(gf + (size_t)(b * Eq + e0) * Gq + istart) = o0;
    *(v8h*)(gf + (size_t)(b * Eq + e1) * Gq + istart) = o1;
}

// ---------------------------------------------------------------------------
// Kernel 2: gene_out[l][m][k] = sum_g gf[m][g]*W_dg[l][k][g] + b_dg[l][k]
// Plain f16 WMMA GEMM: (512 x 2048) @ (2048 x 1024). grid=(32,4), block=256.
// ---------------------------------------------------------------------------
__global__ __launch_bounds__(256) void k_gene(
    const _Float16* __restrict__ gf,    // [512][2048]
    const _Float16* __restrict__ Wdg,   // [1024][2048]  (row = l*64+k)
    const float* __restrict__ b_dg,     // [NL][GS]
    float* __restrict__ out)            // [NL][512][GS]
{
    const int wave = threadIdx.x >> 5;
    const int lane = threadIdx.x & 31;
    const int lm = lane & 15, hi = lane >> 4, base = hi * 8;
    const int mtile = blockIdx.y * 8 + wave;   // 0..31
    const int n0 = blockIdx.x * 32;            // 0..1023 step 32 (never crosses l)
    const _Float16* arow  = gf  + (size_t)(mtile * 16 + lm) * Gq;
    const _Float16* brow0 = Wdg + (size_t)(n0 + lm) * Gq;
    const _Float16* brow1 = Wdg + (size_t)(n0 + 16 + lm) * Gq;
    v8f acc0 = {}, acc1 = {};
    for (int k0 = 0; k0 < Gq; k0 += 32) {
        const v8h* pa = (const v8h*)(arow  + k0 + base);
        const v8h* p0 = (const v8h*)(brow0 + k0 + base);
        const v8h* p1 = (const v8h*)(brow1 + k0 + base);
        v8h alo = pa[0], ahi = pa[2];
        v8h b0lo = p0[0], b0hi = p0[2];
        v8h b1lo = p1[0], b1hi = p1[2];
        v16h a, b0, b1;
        #pragma unroll
        for (int t = 0; t < 8; ++t) {
            a[t]  = alo[t];  a[t + 8]  = ahi[t];
            b0[t] = b0lo[t]; b0[t + 8] = b0hi[t];
            b1[t] = b1lo[t]; b1[t + 8] = b1hi[t];
        }
        acc0 = __builtin_amdgcn_wmma_f32_16x16x32_f16(false, a, false, b0, (short)0, acc0, false, false);
        acc1 = __builtin_amdgcn_wmma_f32_16x16x32_f16(false, a, false, b1, (short)0, acc1, false, false);
    }
    const int col0 = n0 + lm, col1 = col0 + 16;
    const int l = col0 >> 6;
    const int kk0 = col0 & 63, kk1 = col1 & 63;
    const float bias0 = b_dg[l * GSq + kk0], bias1 = b_dg[l * GSq + kk1];
    float* outl = out + (size_t)l * (512 * GSq);
    #pragma unroll
    for (int r = 0; r < 8; ++r) {
        int m = mtile * 16 + r + hi * 8;
        outl[(size_t)m * GSq + kk0] = acc0[r] + bias0;
        outl[(size_t)m * GSq + kk1] = acc1[r] + bias1;
    }
}

// ---------------------------------------------------------------------------
// _term 3D part (h = bn3(ci@W1 + b1); out3d = bn3(h@Wc + bc)). One block / l.
// ---------------------------------------------------------------------------
__device__ inline void bn3_norm(float* buf, int ch, float* red, float* red2,
                                float* stat, int tid) {
    float s = 0.f, ss = 0.f;
    for (int v = tid; v < Bq * Eq; v += 256) {
        int b = v / Eq, e = v % Eq;
        float val = buf[(b * Hq + ch) * Eq + e];
        s += val; ss += val * val;
    }
    red[tid] = s; red2[tid] = ss;
    __syncthreads();
    for (int off = 128; off > 0; off >>= 1) {
        if (tid < off) { red[tid] += red[tid + off]; red2[tid] += red2[tid + off]; }
        __syncthreads();
    }
    if (tid == 0) {
        float m = red[0] / (float)(Bq * Eq);
        stat[0] = m;
        stat[1] = rsqrtf(red2[0] / (float)(Bq * Eq) - m * m + 1e-5f);
    }
    __syncthreads();
    float m = stat[0], inv = stat[1];
    for (int v = tid; v < Bq * Eq; v += 256) {
        int b = v / Eq, e = v % Eq;
        int id = (b * Hq + ch) * Eq + e;
        buf[id] = (buf[id] - m) * inv;
    }
    __syncthreads();
}

__global__ __launch_bounds__(256) void k_term(
    const float* __restrict__ ci,   // [L][B][E][IN]
    const float* __restrict__ W1,   // [L][H][IN]
    const float* __restrict__ b1,   // [L][H]
    const float* __restrict__ Wc,   // [L][H][H]
    const float* __restrict__ bc,   // [L][H]
    float* __restrict__ out3d,      // [L][B][H][E]
    int IN)
{
    __shared__ float hbuf[Bq * Hq * Eq];
    __shared__ float cbuf[Bq * Hq * Eq];
    __shared__ float red[256], red2[256], stat[2];
    const int l = blockIdx.x;
    const int tid = threadIdx.x;
    const float* cil = ci + (size_t)l * Bq * Eq * IN;
    const float* W1l = W1 + (size_t)l * Hq * IN;
    for (int v = tid; v < Bq * Hq * Eq; v += 256) {
        int b = v / (Hq * Eq), h = (v / Eq) % Hq, e = v % Eq;
        const float* row = cil + (size_t)(b * Eq + e) * IN;
        const float* w = W1l + h * IN;
        float s = 0.f;
        for (int i = 0; i < IN; ++i) s += row[i] * w[i];
        hbuf[v] = s + b1[l * Hq + h];
    }
    __syncthreads();
    for (int h = 0; h < Hq; ++h) bn3_norm(hbuf, h, red, red2, stat, tid);
    const float* Wcl = Wc + (size_t)l * Hq * Hq;
    for (int v = tid; v < Bq * Hq * Eq; v += 256) {
        int b = v / (Hq * Eq), k = (v / Eq) % Hq, e = v % Eq;
        float s = 0.f;
        for (int h = 0; h < Hq; ++h) s += hbuf[(b * Hq + h) * Eq + e] * Wcl[k * Hq + h];
        cbuf[v] = s + bc[l * Hq + k];
    }
    __syncthreads();
    for (int k = 0; k < Hq; ++k) bn3_norm(cbuf, k, red, red2, stat, tid);
    for (int v = tid; v < Bq * Hq * Eq; v += 256)
        out3d[(size_t)l * Bq * Hq * Eq + v] = cbuf[v];
}

// ---------------------------------------------------------------------------
// root out2d = bn2(tanh(mean_k(out3d) @ Wd.T + bd))
// ---------------------------------------------------------------------------
__global__ __launch_bounds__(256) void k_root2d(
    const float* __restrict__ out3d,  // [B][H][E]
    const float* __restrict__ Wd,     // [H][E]
    const float* __restrict__ bd,     // [H]
    float* __restrict__ out2d)        // [B][H]
{
    __shared__ float meanbuf[Bq * Eq];
    __shared__ float obuf[Bq * Hq];
    int tid = threadIdx.x;
    for (int v = tid; v < Bq * Eq; v += 256) {
        int b = v / Eq, e = v % Eq;
        float s = 0.f;
        for (int k = 0; k < Hq; ++k) s += out3d[(b * Hq + k) * Eq + e];
        meanbuf[v] = s / (float)Hq;
    }
    __syncthreads();
    for (int v = tid; v < Bq * Hq; v += 256) {
        int b = v / Hq, h = v % Hq;
        float s = 0.f;
        for (int e = 0; e < Eq; ++e) s += meanbuf[b * Eq + e] * Wd[h * Eq + e];
        obuf[v] = tanhf(s + bd[h]);
    }
    __syncthreads();
    if (tid < Hq) {
        float m = 0.f, ss = 0.f;
        for (int b = 0; b < Bq; ++b) { float v = obuf[b * Hq + tid]; m += v; ss += v * v; }
        m /= (float)Bq;
        float inv = rsqrtf(ss / (float)Bq - m * m + 1e-5f);
        for (int b = 0; b < Bq; ++b) out2d[b * Hq + tid] = (obuf[b * Hq + tid] - m) * inv;
    }
}

// ---------------------------------------------------------------------------
// tensor repacks: a[m,b,e,j*6+h] = leaf3d[m*4+j,b,h,e];  r[b,e,j*6+h] = mid3d[j,b,h,e]
// ---------------------------------------------------------------------------
__global__ void k_pack_a(const float* __restrict__ leaf3d, float* __restrict__ a, int n) {
    int o = blockIdx.x * 256 + threadIdx.x;
    if (o >= n) return;
    int c = o % 24; int e = (o / 24) % Eq; int b = (o / (24 * Eq)) % Bq; int m = o / (24 * Eq * Bq);
    int j = c / Hq, h = c % Hq;
    int l = m * 4 + j;
    a[o] = leaf3d[((l * Bq + b) * Hq + h) * Eq + e];
}
__global__ void k_pack_r(const float* __restrict__ mid3d, float* __restrict__ r, int n) {
    int o = blockIdx.x * 256 + threadIdx.x;
    if (o >= n) return;
    int c = o % 24; int e = (o / 24) % Eq; int b = o / (24 * Eq);
    int j = c / Hq, h = c % Hq;
    r[o] = mid3d[((j * Bq + b) * Hq + h) * Eq + e];
}

// ---------------------------------------------------------------------------
// drug MLP layer: out = bn2(tanh(in @ W.T + b)), one block per output column
// ---------------------------------------------------------------------------
__global__ __launch_bounds__(256) void k_fc_bn(
    const float* __restrict__ in, int in_stride, int in_off, int IN,
    const float* __restrict__ W, const float* __restrict__ bias,
    float* __restrict__ out, int OUT)
{
    __shared__ float red[16][17];
    __shared__ float col[16];
    const int c = blockIdx.x;
    const int tid = threadIdx.x;
    const int row = tid >> 4, slot = tid & 15;
    const float* w = W + (size_t)c * IN;
    const float* xr = in + (size_t)row * in_stride + in_off;
    float s = 0.f;
    for (int i = slot; i < IN; i += 16) s += xr[i] * w[i];
    red[row][slot] = s;
    __syncthreads();
    if (tid < 16) {
        float t = 0.f;
        for (int k = 0; k < 16; ++k) t += red[tid][k];
        col[tid] = tanhf(t + bias[c]);
    }
    __syncthreads();
    if (tid == 0) {
        float m = 0.f, ss = 0.f;
        for (int b = 0; b < 16; ++b) { m += col[b]; ss += col[b] * col[b]; }
        m /= 16.f;
        float inv = rsqrtf(ss / 16.f - m * m + 1e-5f);
        for (int b = 0; b < 16; ++b) out[(size_t)b * OUT + c] = (col[b] - m) * inv;
    }
}

// ---------------------------------------------------------------------------
// final head: fin=[root2d|d3]; out=bn2(tanh(fin@Wf.T+bf)); aux=tanh(out@Wfa.T+bfa);
// res = aux*Wfo + bfo
// ---------------------------------------------------------------------------
__global__ __launch_bounds__(64) void k_final(
    const float* __restrict__ root2d, const float* __restrict__ d3,
    const float* __restrict__ Wf, const float* __restrict__ bf,
    const float* __restrict__ Wfa, const float* __restrict__ bfa,
    const float* __restrict__ Wfo, const float* __restrict__ bfo,
    float* __restrict__ res)
{
    __shared__ float fin[16][12];
    __shared__ float obuf[16][6];
    int tid = threadIdx.x;
    for (int v = tid; v < 16 * 12; v += 64) {
        int b = v / 12, c = v % 12;
        fin[b][c] = (c < 6) ? root2d[b * 6 + c] : d3[b * 6 + (c - 6)];
    }
    __syncthreads();
    for (int v = tid; v < 16 * 6; v += 64) {
        int b = v / 6, h = v % 6;
        float s = 0.f;
        for (int i = 0; i < 12; ++i) s += fin[b][i] * Wf[h * 12 + i];
        obuf[b][h] = tanhf(s + bf[h]);
    }
    __syncthreads();
    if (tid < 6) {
        int h = tid;
        float m = 0.f, ss = 0.f;
        for (int b = 0; b < 16; ++b) { float v = obuf[b][h]; m += v; ss += v * v; }
        m /= 16.f;
        float inv = rsqrtf(ss / 16.f - m * m + 1e-5f);
        for (int b = 0; b < 16; ++b) obuf[b][h] = (obuf[b][h] - m) * inv;
    }
    __syncthreads();
    if (tid < 16) {
        int b = tid;
        float s = 0.f;
        for (int h = 0; h < 6; ++h) s += obuf[b][h] * Wfa[h];
        float aux = tanhf(s + bfa[0]);
        res[b] = aux * Wfo[0] + bfo[0];
    }
}

// ---------------------------------------------------------------------------
extern "C" void kernel_launch(void* const* d_in, const int* in_sizes, int n_in,
                              void* d_out, int out_size, void* d_ws, size_t ws_size,
                              hipStream_t stream) {
    (void)in_sizes; (void)n_in; (void)out_size; (void)ws_size;
    const float* x     = (const float*)d_in[0];
    const int*   table = (const int*)  d_in[1];
    const float* W_rbf = (const float*)d_in[2];
    const float* b_rbf = (const float*)d_in[3];
    const float* emb   = (const float*)d_in[4];
    const float* W_dg  = (const float*)d_in[5];
    const float* b_dg  = (const float*)d_in[6];
    const float* Wl1   = (const float*)d_in[7];
    const float* bl1   = (const float*)d_in[8];
    const float* Wlc   = (const float*)d_in[9];
    const float* blc   = (const float*)d_in[10];
    // d_in[11]=Wld, d_in[12]=bld : leaf out2d is discarded by the reference
    const float* Wm1   = (const float*)d_in[13];
    const float* bm1   = (const float*)d_in[14];
    const float* Wmc   = (const float*)d_in[15];
    const float* bmc   = (const float*)d_in[16];
    // d_in[17]=Wmd, d_in[18]=bmd : mid out2d discarded
    const float* Wr1   = (const float*)d_in[19];
    const float* br1   = (const float*)d_in[20];
    const float* Wrc   = (const float*)d_in[21];
    const float* brc   = (const float*)d_in[22];
    const float* Wrd   = (const float*)d_in[23];
    const float* brd   = (const float*)d_in[24];
    const float* Wd1   = (const float*)d_in[25];
    const float* bd1   = (const float*)d_in[26];
    const float* Wd2   = (const float*)d_in[27];
    const float* bd2   = (const float*)d_in[28];
    const float* Wd3   = (const float*)d_in[29];
    const float* bd3   = (const float*)d_in[30];
    const float* Wf    = (const float*)d_in[31];
    const float* bf    = (const float*)d_in[32];
    const float* Wfa   = (const float*)d_in[33];
    const float* bfa   = (const float*)d_in[34];
    const float* Wfo   = (const float*)d_in[35];
    const float* bfo   = (const float*)d_in[36];

    // workspace suballocation (~9 MB total), 256B aligned
    char* p = (char*)d_ws;
    auto alloc = [&](size_t bytes) -> char* {
        char* r = p;
        p += (bytes + 255) & ~(size_t)255;
        return r;
    };
    _Float16* Wh    = (_Float16*)alloc(sizeof(_Float16) * Eq * Gq);
    _Float16* Wdgh  = (_Float16*)alloc(sizeof(_Float16) * NLq * GSq * Gq);
    _Float16* gfh   = (_Float16*)alloc(sizeof(_Float16) * Bq * Eq * Gq);
    float* gene_out = (float*)alloc(sizeof(float) * NLq * Bq * Eq * GSq);
    float* leaf3d   = (float*)alloc(sizeof(float) * NLq * Bq * Hq * Eq);
    float* a_mid    = (float*)alloc(sizeof(float) * NMq * Bq * Eq * 4 * Hq);
    float* mid3d    = (float*)alloc(sizeof(float) * NMq * Bq * Hq * Eq);
    float* r_root   = (float*)alloc(sizeof(float) * Bq * Eq * 4 * Hq);
    float* root3d   = (float*)alloc(sizeof(float) * Bq * Hq * Eq);
    float* root2d   = (float*)alloc(sizeof(float) * Bq * Hq);
    float* dbuf1    = (float*)alloc(sizeof(float) * Bq * 100);
    float* dbuf2    = (float*)alloc(sizeof(float) * Bq * 50);
    float* dbuf3    = (float*)alloc(sizeof(float) * Bq * Hq);

    // stage weights as f16
    k_f2h<<<(Eq * Gq + 255) / 256, 256, 0, stream>>>(W_rbf, Wh, Eq * Gq);
    k_f2h<<<(NLq * GSq * Gq + 255) / 256, 256, 0, stream>>>(W_dg, Wdgh, NLq * GSq * Gq);

    // dominant WMMA kernels
    k_rbf_gv<<<dim3(Gq / 128, Bq), 256, 0, stream>>>(x, Wh, b_rbf, emb, table, gfh);
    k_gene<<<dim3(32, 4), 256, 0, stream>>>(gfh, Wdgh, b_dg, gene_out);

    // tree tail
    k_term<<<NLq, 256, 0, stream>>>(gene_out, Wl1, bl1, Wlc, blc, leaf3d, GSq);
    k_pack_a<<<(NMq * Bq * Eq * 4 * Hq + 255) / 256, 256, 0, stream>>>(leaf3d, a_mid, NMq * Bq * Eq * 4 * Hq);
    k_term<<<NMq, 256, 0, stream>>>(a_mid, Wm1, bm1, Wmc, bmc, mid3d, 4 * Hq);
    k_pack_r<<<(Bq * Eq * 4 * Hq + 255) / 256, 256, 0, stream>>>(mid3d, r_root, Bq * Eq * 4 * Hq);
    k_term<<<1, 256, 0, stream>>>(r_root, Wr1, br1, Wrc, brc, root3d, 4 * Hq);
    k_root2d<<<1, 256, 0, stream>>>(root3d, Wrd, brd, root2d);

    // drug MLP
    k_fc_bn<<<100, 256, 0, stream>>>(x, Gq + Dq + 1, Gq, Dq, Wd1, bd1, dbuf1, 100);
    k_fc_bn<<<50, 256, 0, stream>>>(dbuf1, 100, 0, 100, Wd2, bd2, dbuf2, 50);
    k_fc_bn<<<Hq, 256, 0, stream>>>(dbuf2, 50, 0, 50, Wd3, bd3, dbuf3, Hq);

    // head
    k_final<<<1, 64, 0, stream>>>(root2d, dbuf3, Wf, bf, Wfa, bfa, Wfo, bfo, (float*)d_out);
}
